// Encoder_60112362275055
// MI455X (gfx1250) — matrix operand order repaired and checked by reading, not compiled, not run on hardware
//
#include <hip/hip_runtime.h>
#include <hip/hip_bf16.h>
#include <stdint.h>

// Problem dims (fixed by reference setup_inputs)
#define B_    64
#define T_    512
#define X_    256
#define H_    1024
#define RING  16          // >= 2*skip_size (10); power of two
#define NWG   64          // persistent workgroups, one N-tile (jt) each
#define NTHR  128         // 4 waves of 32; wave mt owns batch rows mt*16..+15
// LDS layout for the workgroup's weight slice (bf16):
//  W_hh: 48 rows (r,z,n x 16) of 1024, padded: +8 bf16 mid-row, +8 end  -> stride 1040
//  W_ih: 48 rows of 256, padded +8 bf16 at end                          -> stride 264
#define SWHH_STRIDE 1040
#define SWIH_STRIDE 264
#define LDS_BYTES ((48 * SWHH_STRIDE + 48 * SWIH_STRIDE) * 2)   // 125184

typedef __attribute__((ext_vector_type(16))) __bf16 v16bf;
typedef __attribute__((ext_vector_type(8)))  float  v8f;
typedef unsigned int v4u __attribute__((ext_vector_type(4)));
typedef int          v8i __attribute__((ext_vector_type(8)));
typedef int          v4i __attribute__((ext_vector_type(4)));

#if defined(__has_builtin)
#  if __has_builtin(__builtin_amdgcn_tensor_load_to_lds) && \
      __has_builtin(__builtin_amdgcn_s_wait_tensorcnt)
#    define HAVE_TDM 1
#  endif
#endif
#ifndef HAVE_TDM
#  define HAVE_TDM 0
#endif

__device__ __forceinline__ v8f wmma_bf16(v16bf a, v16bf b, v8f c) {
  // (neg_a, A, neg_b, B, c_mod, C, reuse_a, reuse_b)
  return __builtin_amdgcn_wmma_f32_16x16x32_bf16(false, a, false, b, (short)0, c,
                                                 false, false);
}

// A-fragment (16x32 bf16) loaded from GLOBAL row-major [rows][kdim]:
// lanes 0-15: row M=lane, K=0..15; lanes 16-31: row M=lane-16, K=16..31
__device__ __forceinline__ v16bf gbl_afrag(const __bf16* base, int row0, int kdim,
                                           int kb, int lane) {
  const int m  = lane & 15;
  const int c0 = kb + ((lane < 16) ? 0 : 16);
  const __bf16* p = base + (size_t)(row0 + m) * kdim + c0;
  v16bf a;
  ((uint4*)&a)[0] = *(const uint4*)(const void*)(p);
  ((uint4*)&a)[1] = *(const uint4*)(const void*)(p + 8);
  return a;
}

// B-fragment (32x16 bf16) from LDS weight slice; `cc` is the padded column offset.
// lanes 0-15: N=lane, K=kb..kb+15 ; lanes 16-31: N=lane-16, K=kb+16..kb+31
__device__ __forceinline__ v16bf lds_bfrag(const __bf16* w, int row0, int stride,
                                           int cc, int lane) {
  const __bf16* p = w + (row0 + (lane & 15)) * stride + cc;
  v16bf b;
  ((uint4*)&b)[0] = *(const uint4*)(const void*)(p);
  ((uint4*)&b)[1] = *(const uint4*)(const void*)(p + 8);
  return b;
}

__device__ __forceinline__ float sigm(float v) { return 1.0f / (1.0f + __expf(-v)); }

#if HAVE_TDM
// One-shot TDM: 2D tile (rows x rowlen bf16) global -> LDS, with LDS padding.
// pad_icode: 6 -> every 128 dwords, 7 -> every 256 dwords; pad amount 4 dwords (8 bf16).
__device__ __forceinline__ void tdm_load_2d(const __bf16* gsrc, unsigned lds_off,
                                            int rows, int rowlen, int pad_icode) {
  unsigned long long ga = (unsigned long long)(uintptr_t)(const void*)gsrc;
  v4u g0;
  g0[0] = 1u;                                   // count=1, user descriptor
  g0[1] = lds_off;                              // lds_addr (bytes)
  g0[2] = (unsigned)(ga & 0xffffffffu);         // global_addr[31:0]
  g0[3] = (unsigned)((ga >> 32) & 0x1ffffffu) | (2u << 30);  // addr[56:32] | type=2
  const int dim0 = rowlen, dim1 = 4096, tile0 = rowlen, tile1 = rows;
  v8i g1;
  g1[0] = (1 << 16) | (1 << 20) | (pad_icode << 22) | (3 << 25); // 2B, pad_en, amt=4dw
  g1[1] = (dim0 & 0xffff) << 16;
  g1[2] = ((dim0 >> 16) & 0xffff) | ((dim1 & 0xffff) << 16);
  g1[3] = ((dim1 >> 16) & 0xffff) | ((tile0 & 0xffff) << 16);
  g1[4] = (tile1 & 0xffff);                     // tile_dim2 = 0
  g1[5] = rowlen;                               // tensor_dim0_stride[31:0]
  g1[6] = 0;
  g1[7] = 0;
  v4i z4 = {0, 0, 0, 0};
#if __clang_major__ >= 23
  v8i z8 = {0, 0, 0, 0, 0, 0, 0, 0};
  __builtin_amdgcn_tensor_load_to_lds(g0, g1, z4, z4, z8, 0);
#else
  __builtin_amdgcn_tensor_load_to_lds(g0, g1, z4, z4, 0);
#endif
}
#endif

__global__ void init_kernel(int* arrive, __bf16* hbuf0) {
  const int i = blockIdx.x * blockDim.x + threadIdx.x;
  if (i < T_) arrive[i] = 0;
  for (int j = i; j < B_ * H_; j += gridDim.x * blockDim.x) hbuf0[j] = (__bf16)0.0f;
}

__global__ void cvt_bf16_kernel(const float* __restrict__ src,
                                __bf16* __restrict__ dst, int n) {
  for (int i = blockIdx.x * blockDim.x + threadIdx.x; i < n;
       i += gridDim.x * blockDim.x)
    dst[i] = (__bf16)src[i];
}

// x [B][T][X] fp32 -> xbf [T][B][X] bf16 (step-t A-tiles contiguous)
__global__ void cvt_x_kernel(const float* __restrict__ src,
                             __bf16* __restrict__ dst) {
  for (size_t i = (size_t)blockIdx.x * blockDim.x + threadIdx.x;
       i < (size_t)B_ * T_ * X_; i += (size_t)gridDim.x * blockDim.x) {
    const int    c  = (int)(i & (X_ - 1));
    const size_t bt = i >> 8;                   // / X_
    const int    t  = (int)(bt & (T_ - 1));
    const int    b  = (int)(bt >> 9);           // / T_
    dst[((size_t)t * B_ + b) * X_ + c] = (__bf16)src[i];
  }
}

__global__ void __launch_bounds__(NTHR, 1)
rnn_persistent_kernel(const __bf16* __restrict__ xbf,   // [T][B][X] bf16
                      const float*  __restrict__ b_ih,
                      const float*  __restrict__ b_hh,
                      const int*    __restrict__ w1,
                      const int*    __restrict__ w2,
                      const int*    __restrict__ skip_p,
                      const __bf16* __restrict__ Whh,   // [3H][H] bf16
                      const __bf16* __restrict__ Wih,   // [3H][X] bf16
                      float*  __restrict__ ring,        // [RING][B][H] fp32
                      __bf16* __restrict__ hbuf,        // [2][B][H] bf16 (A-matrix)
                      int*    __restrict__ arrive,      // [T]
                      float*  __restrict__ out)         // [B][H] fp32
{
  extern __shared__ __bf16 smem[];
  __bf16* sWhh = smem;                      // [48][SWHH_STRIDE]
  __bf16* sWih = smem + 48 * SWHH_STRIDE;   // [48][SWIH_STRIDE]

  const int tid  = threadIdx.x;
  const int lane = tid & 31;
  const int wave = tid >> 5;
  const int jt   = blockIdx.x;              // N tile (columns jt*16..+15 of h)
  const int mt   = wave;                    // M tile (batch rows mt*16..+15)
  const int jcol = jt * 16 + (lane & 15);
  const int sk   = skip_p[0];

  const int grow[3] = {jt * 16, H_ + jt * 16, 2 * H_ + jt * 16};  // r, z, n rows

  // ---- one-time: stage this WG's weight slice into LDS ----
#if HAVE_TDM
  if (wave == 0) {
    #pragma unroll
    for (int g = 0; g < 3; ++g) {
      const unsigned ldsW = (unsigned)(uintptr_t)(void*)(sWhh + g * 16 * SWHH_STRIDE);
      const unsigned ldsI = (unsigned)(uintptr_t)(void*)(sWih + g * 16 * SWIH_STRIDE);
      tdm_load_2d(Whh + (size_t)grow[g] * H_, ldsW, 16, H_, 7);  // pad every 256 dw
      tdm_load_2d(Wih + (size_t)grow[g] * X_, ldsI, 16, X_, 6);  // pad every 128 dw
    }
    __builtin_amdgcn_s_wait_tensorcnt(0);
  }
#else
  for (int i = tid; i < 48 * H_; i += NTHR) {    // manual fallback, same layout
    const int r = i >> 10, c = i & (H_ - 1);
    sWhh[r * SWHH_STRIDE + c + (((unsigned)c >> 9) << 3)] =
        Whh[(size_t)(grow[r >> 4] + (r & 15)) * H_ + c];
  }
  for (int i = tid; i < 48 * X_; i += NTHR) {
    const int r = i >> 8, c = i & (X_ - 1);
    sWih[r * SWIH_STRIDE + c] = Wih[(size_t)(grow[r >> 4] + (r & 15)) * X_ + c];
  }
#endif
  __syncthreads();

  // Per-lane biases, time-invariant
  const float br  = b_ih[jcol]          + b_hh[jcol];
  const float bz  = b_ih[H_ + jcol]     + b_hh[H_ + jcol];
  const float bxn = b_ih[2 * H_ + jcol];
  const float bhn = b_hh[2 * H_ + jcol];
  const int   mhi = (lane >> 4) << 3;

  for (int t = 0; t < T_; ++t) {
    // ---- grid barrier: all of h_comb(t) (written at step t-1) is visible ----
    if (t > 0) {
      if (tid == 0) {
        while (__hip_atomic_load(&arrive[t - 1], __ATOMIC_ACQUIRE,
                                 __HIP_MEMORY_SCOPE_AGENT) < NWG)
          __builtin_amdgcn_s_sleep(1);
      }
      __syncthreads();
    }

    const __bf16* hb = hbuf + (size_t)(t & 1) * (B_ * H_);   // combined h (A-matrix)
    const __bf16* xb = xbf  + (size_t)t * (B_ * X_);

    v8f accr  = {0.f, 0.f, 0.f, 0.f, 0.f, 0.f, 0.f, 0.f};
    v8f accz  = accr, acchn = accr, accxn = accr;

    #pragma unroll 4
    for (int kb = 0; kb < H_; kb += 32) {        // h contribution (K=1024)
      v16bf af = gbl_afrag(hb, mt * 16, H_, kb, lane);
      int c0 = kb + ((lane < 16) ? 0 : 16);
      int cc = c0 + (((unsigned)c0 >> 9) << 3);  // +8 bf16 mid-row pad
      v16bf b0 = lds_bfrag(sWhh,  0, SWHH_STRIDE, cc, lane);
      v16bf b1 = lds_bfrag(sWhh, 16, SWHH_STRIDE, cc, lane);
      v16bf b2 = lds_bfrag(sWhh, 32, SWHH_STRIDE, cc, lane);
      accr  = wmma_bf16(af, b0, accr);
      accz  = wmma_bf16(af, b1, accz);
      acchn = wmma_bf16(af, b2, acchn);
    }
    #pragma unroll 4
    for (int kb = 0; kb < X_; kb += 32) {        // x contribution (K=256)
      v16bf af = gbl_afrag(xb, mt * 16, X_, kb, lane);
      int cc = kb + ((lane < 16) ? 0 : 16);
      v16bf b0 = lds_bfrag(sWih,  0, SWIH_STRIDE, cc, lane);
      v16bf b1 = lds_bfrag(sWih, 16, SWIH_STRIDE, cc, lane);
      v16bf b2 = lds_bfrag(sWih, 32, SWIH_STRIDE, cc, lane);
      accr  = wmma_bf16(af, b0, accr);
      accz  = wmma_bf16(af, b1, accz);
      accxn = wmma_bf16(af, b2, accxn);
    }

    // ---- skip plan for step t+1 (scalar; replicates _skip_plan) ----
    const int t2 = t + 1;
    float w1n = 0.f, w2n = 0.f;
    int   sidx2 = 0;
    bool  uz2 = true;
    if (t2 < T_) {
      w1n = (float)w1[t2];
      w2n = (float)w2[t2];
      if (t2 < sk) {
        if (2 * t2 >= sk) { uz2 = false; sidx2 = 2 * t2 - sk; }
      } else {
        if (t2 - sk >= sk) { uz2 = false; sidx2 = t2 - 2 * sk; }
      }
    }
    const float* ring_skip = ring + (size_t)(sidx2 & (RING - 1)) * (B_ * H_);
    float*  ring_t = ring + (size_t)(t & (RING - 1)) * (B_ * H_);
    __bf16* hb_nxt = hbuf + (size_t)(t2 & 1) * (B_ * H_);

    // ---- gates + fused writeback of h_t (ring) and h_comb(t+1) (bf16 A) ----
    #pragma unroll
    for (int v = 0; v < 8; ++v) {
      const int m = mt * 16 + v + mhi;           // batch row 0..63
      const size_t e = (size_t)m * H_ + jcol;
      float r  = sigm(accr[v] + br);
      float z  = sigm(accz[v] + bz);
      float nn = tanhf(accxn[v] + bxn + r * (acchn[v] + bhn));
      float hv = (float)hb[e];                   // combined h (pre-gate)
      float hn2 = (1.0f - z) * nn + z * hv;
      ring_t[e] = hn2;
      if (t == T_ - 1) {
        out[e] = 2.0f * hn2;
      } else {
        float hs2 = uz2 ? 0.0f : ((sidx2 == t) ? hn2 : ring_skip[e]);
        hb_nxt[e] = (__bf16)(w1n * hn2 + w2n * hs2);
      }
    }

    __threadfence();
    __syncthreads();
    if (tid == 0) {
      __hip_atomic_fetch_add(&arrive[t], 1, __ATOMIC_RELEASE,
                             __HIP_MEMORY_SCOPE_AGENT);
    }
  }
}

extern "C" void kernel_launch(void* const* d_in, const int* in_sizes, int n_in,
                              void* d_out, int out_size, void* d_ws, size_t ws_size,
                              hipStream_t stream) {
  const float* x    = (const float*)d_in[0];   // [B,T,X]
  const float* W_ih = (const float*)d_in[1];   // [3H,X]
  const float* W_hh = (const float*)d_in[2];   // [3H,H]
  const float* b_ih = (const float*)d_in[3];   // [3H]
  const float* b_hh = (const float*)d_in[4];   // [3H]
  const int*   w1   = (const int*)d_in[5];     // [T]
  const int*   w2   = (const int*)d_in[6];     // [T]
  const int*   skip = (const int*)d_in[7];     // scalar
  float*       out  = (float*)d_out;           // [B,H]

  char* ws = (char*)d_ws;
  size_t off = 0;
  int*    arrive = (int*)(ws + off);    off += 4096;
  __bf16* whh    = (__bf16*)(ws + off); off += (size_t)3 * H_ * H_ * 2;   // 6 MB
  __bf16* wih    = (__bf16*)(ws + off); off += (size_t)3 * H_ * X_ * 2;   // 1.5 MB
  float*  ring   = (float*)(ws + off);  off += (size_t)RING * B_ * H_ * 4; // 4 MB
  __bf16* hbuf   = (__bf16*)(ws + off); off += (size_t)2 * B_ * H_ * 2;   // 256 KB
  __bf16* xbf    = (__bf16*)(ws + off); off += (size_t)T_ * B_ * X_ * 2;  // 16 MB

  init_kernel<<<64, 256, 0, stream>>>(arrive, hbuf);
  cvt_bf16_kernel<<<1024, 256, 0, stream>>>(W_hh, whh, 3 * H_ * H_);
  cvt_bf16_kernel<<<256, 256, 0, stream>>>(W_ih, wih, 3 * H_ * X_);
  cvt_x_kernel<<<2048, 256, 0, stream>>>(x, xbf);

  (void)hipFuncSetAttribute((const void*)rnn_persistent_kernel,
                            hipFuncAttributeMaxDynamicSharedMemorySize,
                            (int)LDS_BYTES);
  rnn_persistent_kernel<<<NWG, NTHR, LDS_BYTES, stream>>>(
      xbf, b_ih, b_hh, w1, w2, skip, whh, wih, ring, hbuf, arrive, out);
}
